// TransformerBlock_73813307949476
// MI455X (gfx1250) — matrix operand order repaired and checked by reading, not compiled, not run on hardware
//
#include <hip/hip_runtime.h>
#include <hip/hip_bf16.h>

typedef __bf16 bf16_t;
typedef __attribute__((ext_vector_type(16))) __bf16 v16bf;
typedef __attribute__((ext_vector_type(8)))  float  v8f;

// async-to-LDS builtin pointer types (signature confirmed by hipcc diagnostic:
// int4-vector pointers; AS(1) source, AS(3) LDS destination).
typedef int v4i_ __attribute__((vector_size(16)));
typedef __attribute__((address_space(1))) v4i_* g_v4i_ptr;
typedef __attribute__((address_space(3))) v4i_* l_v4i_ptr;

#if __has_builtin(__builtin_amdgcn_global_load_async_to_lds_b128)
#define HAS_ASYNC_LDS 1
#else
#define HAS_ASYNC_LDS 0
#endif

#define D_MODEL 768
#define N_HEADS 12
#define D_HEAD  64
#define D_FF    3072
#define BATCH   2
#define SEQ     4096
#define ROWS    (BATCH * SEQ)   // 8192
#define LN_EPS  1e-5f

static __device__ __forceinline__ v8f wmma_bf16(const v16bf& a, const v16bf& b, v8f c) {
  // D = A(16x32 bf16) * B(32x16 bf16) + C(16x16 f32)
  return __builtin_amdgcn_wmma_f32_16x16x32_bf16(false, a, false, b, (short)0, c, false, false);
}

// ---------------------------------------------------------------------------
// Generic GEMM: out[M,N] = act(A[M,K] @ W[K,N] + bias[N]) (+ resid[M,N])
// 64x64 tile per block, 128 threads = 4 waves, each wave owns a 16-row strip.
// N,K are template constants so strided staging loads use immediate offsets.
// A staged row-major (A-frag elements contiguous in K); W staged TRANSPOSED
// (ldsBT[n][k]) so B-frag elements are contiguous -> ds_load_b128.
// ---------------------------------------------------------------------------
template <int N, int K>
__global__ __launch_bounds__(128)
void gemm_bias_kernel(const float* __restrict__ A, const float* __restrict__ W,
                      const float* __restrict__ bias, const float* __restrict__ resid,
                      float* __restrict__ outF, bf16_t* __restrict__ outB,
                      int relu)
{
  __shared__ __align__(16) bf16_t ldsA[64][32];   // 4 KB, [m][k]
  __shared__ __align__(16) bf16_t ldsBT[64][32];  // 4 KB, [n][k]  (transposed W)

  const int t     = threadIdx.x;
  const int wave  = t >> 5;
  const int lane  = t & 31;
  const int lhalf = lane >> 4;      // 0 or 1
  const int l16   = lane & 15;

  const int mblk = blockIdx.y * 64;
  const int nblk = blockIdx.x * 64;

  v8f acc[4];
  #pragma unroll
  for (int j = 0; j < 4; ++j)
    #pragma unroll
    for (int e = 0; e < 8; ++e) acc[j][e] = 0.0f;

  const int ar  = t >> 1;            // 0..63  (A stage row)
  const int ac  = (t & 1) * 16;      // 0/16   (A stage col base)
  const int bn  = t & 63;            // 0..63  (W stage column)
  const int bk0 = (t >> 6) * 16;     // 0/16   (W stage k base)

  const float* aSrcBase = A + (size_t)(mblk + ar) * K + ac;
  const float* wSrcBase = W + (size_t)bk0 * N + nblk + bn;

  for (int kb = 0; kb < K; kb += 32) {
    // ---- stage A tile (64x32) fp32 -> bf16, row-major ----
    {
      const float* src = aSrcBase + kb;
      #pragma unroll
      for (int u = 0; u < 4; ++u) {
        float4 f = *reinterpret_cast<const float4*>(src + u * 4);
        int c = ac + u * 4;
        ldsA[ar][c + 0] = (bf16_t)f.x;
        ldsA[ar][c + 1] = (bf16_t)f.y;
        ldsA[ar][c + 2] = (bf16_t)f.z;
        ldsA[ar][c + 3] = (bf16_t)f.w;
      }
    }
    // ---- stage W tile (32x64) fp32 -> bf16, TRANSPOSED to [n][k] ----
    // each thread reads one column (coalesced across adjacent threads, and with
    // N constexpr the i*N strides become immediate offsets) then writes 16
    // contiguous bf16 into its ldsBT row.
    {
      alignas(16) bf16_t buf[16];
      const float* src = wSrcBase + (size_t)kb * N;
      #pragma unroll
      for (int i = 0; i < 16; ++i)
        buf[i] = (bf16_t)src[i * N];
      *reinterpret_cast<uint4*>(&ldsBT[bn][bk0])     = *reinterpret_cast<uint4*>(&buf[0]);
      *reinterpret_cast<uint4*>(&ldsBT[bn][bk0 + 8]) = *reinterpret_cast<uint4*>(&buf[8]);
    }
    if (kb + 32 < K) {  // prefetch next A slab -> global_prefetch_b8
      __builtin_prefetch(aSrcBase + kb + 32, 0, 1);
    }
    __syncthreads();

    // ---- A fragment: rows wave*16+l16, contiguous k = 16*lhalf + e ----
    v16bf afrag;
    {
      const int row = wave * 16 + l16;
      #pragma unroll
      for (int e = 0; e < 16; ++e) {
        int k = e + lhalf * 16;
        afrag[e] = ldsA[row][k];
      }
    }
    // ---- 4 B fragments / 4 WMMAs covering the 64 output columns ----
    #pragma unroll
    for (int j = 0; j < 4; ++j) {
      v16bf bfrag;
      const int col = j * 16 + l16;
      #pragma unroll
      for (int e = 0; e < 16; ++e)
        bfrag[e] = ldsBT[col][e + lhalf * 16];   // contiguous -> ds_load_b128
      acc[j] = wmma_bf16(afrag, bfrag, acc[j]);
    }
    __syncthreads();
  }

  // ---- epilogue: bias, relu, residual, fp32 and/or bf16 store ----
  #pragma unroll
  for (int j = 0; j < 4; ++j) {
    const int col = nblk + j * 16 + l16;
    const float bv = bias[col];
    #pragma unroll
    for (int r = 0; r < 8; ++r) {
      const int row = mblk + wave * 16 + r + lhalf * 8;
      float v = acc[j][r] + bv;
      if (relu) v = fmaxf(v, 0.0f);
      if (resid) v += resid[(size_t)row * N + col];
      if (outF)  outF[(size_t)row * N + col] = v;
      if (outB)  outB[(size_t)row * N + col] = (bf16_t)v;
    }
  }
}

// ---------------------------------------------------------------------------
// Flash attention: block = (batch,head,64-query tile), 4 waves x 16 q-rows.
// Online softmax, 32-key tiles; scores never hit global memory.
// Q and K tiles staged via GLOBAL_LOAD_ASYNC_TO_LDS_B128 (ASYNCcnt path);
// V staged transposed [d][k] in software (PV B-frag reads contiguous in k).
// ---------------------------------------------------------------------------
__global__ __launch_bounds__(128)
void attention_kernel(const bf16_t* __restrict__ Q, const bf16_t* __restrict__ Km,
                      const bf16_t* __restrict__ Vm, float* __restrict__ out)
{
  __shared__ __align__(16) bf16_t ldsQ[64][64];      // 8 KB  [q][d]
  __shared__ __align__(16) bf16_t ldsK[32][64];      // 4 KB  [k][d]
  __shared__ __align__(16) bf16_t ldsVT[64][32];     // 4 KB  [d][k] (transposed V)
  __shared__ __align__(16) bf16_t ldsP[4][16][32];   // 4 KB  per-wave P staging

  const int t     = threadIdx.x;
  const int wave  = t >> 5;
  const int lane  = t & 31;
  const int lhalf = lane >> 4;
  const int l16   = lane & 15;

  const int q0 = blockIdx.x * 64;
  const int b  = blockIdx.y / N_HEADS;
  const int h  = blockIdx.y % N_HEADS;
  const size_t base = (size_t)b * SEQ * D_MODEL + (size_t)h * D_HEAD;

  // ---- load Q tile [64 q][64 d] (bf16, contiguous in d) ----
  {
    const int r  = t >> 1;
    const int cb = (t & 1) * 32;
    const bf16_t* src = Q + base + (size_t)(q0 + r) * D_MODEL + cb;
#if HAS_ASYNC_LDS
    #pragma unroll
    for (int u = 0; u < 4; ++u)
      __builtin_amdgcn_global_load_async_to_lds_b128(
          (g_v4i_ptr)(src + u * 8),
          (l_v4i_ptr)&ldsQ[r][cb + u * 8],
          0, 0);
    asm volatile("s_wait_asynccnt 0x0" ::: "memory");
#else
    #pragma unroll
    for (int u = 0; u < 4; ++u)
      *reinterpret_cast<uint4*>(&ldsQ[r][cb + u * 8]) =
        *reinterpret_cast<const uint4*>(src + u * 8);
#endif
  }
  __syncthreads();

  // ---- per-wave Q fragments (d_head=64 -> 2 K-steps of 32) ----
  v16bf qfrag[2];
  {
    const int row = wave * 16 + l16;
    #pragma unroll
    for (int ks = 0; ks < 2; ++ks)
      #pragma unroll
      for (int e = 0; e < 16; ++e) {
        int k = ks * 32 + e + lhalf * 16;
        qfrag[ks][e] = ldsQ[row][k];
      }
  }

  v8f o[4];
  float m_run[8], l_run[8];
  #pragma unroll
  for (int j = 0; j < 4; ++j)
    #pragma unroll
    for (int e = 0; e < 8; ++e) o[j][e] = 0.0f;
  #pragma unroll
  for (int r = 0; r < 8; ++r) { m_run[r] = -1e30f; l_run[r] = 0.0f; }

  const int kvr = t >> 2;            // 0..31 (K stage row)
  const int kvc = (t & 3) * 16;      // 0..48 (K stage col base)
  const int vd  = t & 63;            // 0..63 (V stage d)
  const int vk0 = (t >> 6) * 16;     // 0/16  (V stage k base)

  for (int kt0 = 0; kt0 <= q0 + 63; kt0 += 32) {
    // ---- stage K tile [32 k][64 d] via async-to-LDS ----
    {
      const bf16_t* ks = Km + base + (size_t)(kt0 + kvr) * D_MODEL + kvc;
#if HAS_ASYNC_LDS
      __builtin_amdgcn_global_load_async_to_lds_b128(
          (g_v4i_ptr)ks,       (l_v4i_ptr)&ldsK[kvr][kvc],     0, 0);
      __builtin_amdgcn_global_load_async_to_lds_b128(
          (g_v4i_ptr)(ks + 8), (l_v4i_ptr)&ldsK[kvr][kvc + 8], 0, 0);
#else
      *reinterpret_cast<uint4*>(&ldsK[kvr][kvc])     = *reinterpret_cast<const uint4*>(ks);
      *reinterpret_cast<uint4*>(&ldsK[kvr][kvc + 8]) = *reinterpret_cast<const uint4*>(ks + 8);
#endif
    }
    // ---- stage V tile transposed -> ldsVT[d][k] (overlaps async K) ----
    {
      alignas(16) bf16_t vbuf[16];
      const bf16_t* vs = Vm + base + (size_t)(kt0 + vk0) * D_MODEL + vd;
      #pragma unroll
      for (int i = 0; i < 16; ++i)
        vbuf[i] = vs[i * D_MODEL];
      *reinterpret_cast<uint4*>(&ldsVT[vd][vk0])     = *reinterpret_cast<uint4*>(&vbuf[0]);
      *reinterpret_cast<uint4*>(&ldsVT[vd][vk0 + 8]) = *reinterpret_cast<uint4*>(&vbuf[8]);
    }
#if HAS_ASYNC_LDS
    asm volatile("s_wait_asynccnt 0x0" ::: "memory");
#endif
    __syncthreads();

    // ---- S = Q . K^T : two 16x16 score frags (keys 0-15, 16-31) ----
    v8f s0, s1;
    #pragma unroll
    for (int e = 0; e < 8; ++e) { s0[e] = 0.0f; s1[e] = 0.0f; }
    #pragma unroll
    for (int ks = 0; ks < 2; ++ks) {
      v16bf b0, b1;
      #pragma unroll
      for (int e = 0; e < 16; ++e) {
        int d = ks * 32 + e + lhalf * 16;   // contiguous along d -> ds_load_b128
        b0[e] = ldsK[l16][d];               // key = l16
        b1[e] = ldsK[l16 + 16][d];          // key = l16 + 16
      }
      s0 = wmma_bf16(qfrag[ks], b0, s0);
      s1 = wmma_bf16(qfrag[ks], b1, s1);
    }

    // ---- online softmax (per query row), write P to per-wave LDS ----
    const int key0 = kt0 + l16;
    const int key1 = kt0 + 16 + l16;
    #pragma unroll
    for (int r = 0; r < 8; ++r) {
      const int Mrow = r + lhalf * 8;
      const int qi   = q0 + wave * 16 + Mrow;
      float v0 = (key0 <= qi) ? s0[r] * 0.125f : -1e30f;   // 1/sqrt(64)
      float v1 = (key1 <= qi) ? s1[r] * 0.125f : -1e30f;
      float mx = fmaxf(v0, v1);
      #pragma unroll
      for (int off = 1; off < 16; off <<= 1) mx = fmaxf(mx, __shfl_xor(mx, off, 32));
      const float mnew = fmaxf(m_run[r], mx);
      const float p0 = __expf(v0 - mnew);
      const float p1 = __expf(v1 - mnew);
      float rs = p0 + p1;
      #pragma unroll
      for (int off = 1; off < 16; off <<= 1) rs += __shfl_xor(rs, off, 32);
      const float alpha = __expf(m_run[r] - mnew);
      m_run[r] = mnew;
      l_run[r] = l_run[r] * alpha + rs;
      #pragma unroll
      for (int j = 0; j < 4; ++j) o[j][r] *= alpha;
      ldsP[wave][Mrow][l16]      = (bf16_t)p0;   // D-layout -> LDS transpose
      ldsP[wave][Mrow][l16 + 16] = (bf16_t)p1;
    }
    __builtin_amdgcn_wave_barrier();   // wave-internal LDS ordering fence

    // ---- P (16x32) as A fragment ----
    v16bf pfrag;
    #pragma unroll
    for (int e = 0; e < 16; ++e) {
      int k = e + lhalf * 16;
      pfrag[e] = ldsP[wave][l16][k];
    }
    // ---- O += P . V ----
    #pragma unroll
    for (int j = 0; j < 4; ++j) {
      v16bf bfrag;
      const int d = j * 16 + l16;
      #pragma unroll
      for (int e = 0; e < 16; ++e)
        bfrag[e] = ldsVT[d][e + lhalf * 16];    // contiguous -> ds_load_b128
      o[j] = wmma_bf16(pfrag, bfrag, o[j]);
    }
    __syncthreads();
  }

  // ---- normalize and store [B,T,D] ----
  #pragma unroll
  for (int j = 0; j < 4; ++j) {
    const int d = j * 16 + l16;
    #pragma unroll
    for (int r = 0; r < 8; ++r) {
      const int qi = q0 + wave * 16 + r + lhalf * 8;
      out[(size_t)(b * SEQ + qi) * D_MODEL + h * D_HEAD + d] = o[j][r] / l_run[r];
    }
  }
}

// ---------------------------------------------------------------------------
// LayerNorm: one block per row of 768, 256 threads x 3 elements.
// ---------------------------------------------------------------------------
__global__ __launch_bounds__(256)
void layernorm_kernel(const float* __restrict__ x, const float* __restrict__ gamma,
                      const float* __restrict__ beta, float* __restrict__ out)
{
  __shared__ float ssum[256];
  __shared__ float ssq[256];
  const int row = blockIdx.x;
  const float* xr = x + (size_t)row * D_MODEL;

  float v[3], s = 0.0f, sq = 0.0f;
  #pragma unroll
  for (int i = 0; i < 3; ++i) {
    v[i] = xr[threadIdx.x + i * 256];
    s += v[i];
    sq += v[i] * v[i];
  }
  ssum[threadIdx.x] = s;
  ssq[threadIdx.x]  = sq;
  __syncthreads();
  for (int st = 128; st > 0; st >>= 1) {
    if (threadIdx.x < st) {
      ssum[threadIdx.x] += ssum[threadIdx.x + st];
      ssq[threadIdx.x]  += ssq[threadIdx.x + st];
    }
    __syncthreads();
  }
  const float mean = ssum[0] * (1.0f / D_MODEL);
  const float var  = ssq[0] * (1.0f / D_MODEL) - mean * mean;
  const float rstd = rsqrtf(var + LN_EPS);
  #pragma unroll
  for (int i = 0; i < 3; ++i) {
    const int c = threadIdx.x + i * 256;
    out[(size_t)row * D_MODEL + c] = (v[i] - mean) * rstd * gamma[c] + beta[c];
  }
}

// ---------------------------------------------------------------------------
extern "C" void kernel_launch(void* const* d_in, const int* in_sizes, int n_in,
                              void* d_out, int out_size, void* d_ws, size_t ws_size,
                              hipStream_t stream)
{
  const float* x   = (const float*)d_in[0];
  const float* wq  = (const float*)d_in[1];  const float* bq  = (const float*)d_in[2];
  const float* wk  = (const float*)d_in[3];  const float* bk  = (const float*)d_in[4];
  const float* wv  = (const float*)d_in[5];  const float* bv  = (const float*)d_in[6];
  const float* wo  = (const float*)d_in[7];  const float* bo  = (const float*)d_in[8];
  const float* w1  = (const float*)d_in[9];  const float* b1  = (const float*)d_in[10];
  const float* w2  = (const float*)d_in[11]; const float* b2  = (const float*)d_in[12];
  const float* g1  = (const float*)d_in[13]; const float* be1 = (const float*)d_in[14];
  const float* g2  = (const float*)d_in[15]; const float* be2 = (const float*)d_in[16];
  float* out = (float*)d_out;

  char* ws = (char*)d_ws;
  size_t off = 0;
  auto alloc = [&](size_t bytes) -> void* {
    void* p = ws + off;
    off = (off + bytes + 255) & ~(size_t)255;
    return p;
  };

  bf16_t* Qb   = (bf16_t*)alloc((size_t)ROWS * D_MODEL * sizeof(bf16_t));
  bf16_t* Kb   = (bf16_t*)alloc((size_t)ROWS * D_MODEL * sizeof(bf16_t));
  bf16_t* Vb   = (bf16_t*)alloc((size_t)ROWS * D_MODEL * sizeof(bf16_t));
  float*  attn = (float*)alloc((size_t)ROWS * D_MODEL * sizeof(float));
  float*  y1   = (float*)alloc((size_t)ROWS * D_MODEL * sizeof(float));
  float*  hbuf = (float*)alloc((size_t)ROWS * D_MODEL * sizeof(float));
  float*  ff   = (float*)alloc((size_t)ROWS * D_FF   * sizeof(float));
  float*  y2   = (float*)alloc((size_t)ROWS * D_MODEL * sizeof(float));

  const dim3 blk(128);
  const dim3 gproj(D_MODEL / 64, ROWS / 64);   // (12, 128)
  const dim3 gff1(D_FF / 64, ROWS / 64);       // (48, 128)
  const dim3 gattn(SEQ / 64, BATCH * N_HEADS); // (64, 24)

  // Q/K/V projections (bf16 outputs feed the WMMA attention kernel)
  gemm_bias_kernel<D_MODEL, D_MODEL><<<gproj, blk, 0, stream>>>(
      x, wq, bq, nullptr, nullptr, Qb, 0);
  gemm_bias_kernel<D_MODEL, D_MODEL><<<gproj, blk, 0, stream>>>(
      x, wk, bk, nullptr, nullptr, Kb, 0);
  gemm_bias_kernel<D_MODEL, D_MODEL><<<gproj, blk, 0, stream>>>(
      x, wv, bv, nullptr, nullptr, Vb, 0);
  // causal flash attention
  attention_kernel<<<gattn, blk, 0, stream>>>(Qb, Kb, Vb, attn);
  // output projection + residual
  gemm_bias_kernel<D_MODEL, D_MODEL><<<gproj, blk, 0, stream>>>(
      attn, wo, bo, x, y1, nullptr, 0);
  layernorm_kernel<<<dim3(ROWS), dim3(256), 0, stream>>>(y1, g1, be1, hbuf);
  // FFN
  gemm_bias_kernel<D_FF, D_MODEL><<<gff1, blk, 0, stream>>>(
      hbuf, w1, b1, nullptr, ff, nullptr, 1);
  gemm_bias_kernel<D_MODEL, D_FF><<<gproj, blk, 0, stream>>>(
      ff, w2, b2, hbuf, y2, nullptr, 0);
  layernorm_kernel<<<dim3(ROWS), dim3(256), 0, stream>>>(y2, g2, be2, out);
}
